// CAPEnsemble_46480136077367
// MI455X (gfx1250) — matrix operand order repaired and checked by reading, not compiled, not run on hardware
//
#include <hip/hip_runtime.h>
#include <hip/hip_bf16.h>
#include <stdint.h>

#define NEGV    (-1e9f)
#define PAD_T    0
#define START_T  1
#define END_T    2
#define STOPLO   3
#define STOPHI   50
#define BADLO    50
#define BADHI    100

constexpr int Mn = 2;      // ensemble models
constexpr int Vn = 10000;  // vocab
constexpr int Hn = 512;
constexpr int En = 512;
constexpr int An = 512;
constexpr int Rn = 36;
constexpr int Bn = 5;      // beam
constexpr int Ln = 30;     // max_len

typedef __attribute__((ext_vector_type(16))) __bf16       v16bf;
typedef __attribute__((ext_vector_type(8)))  float        v8f;
typedef __attribute__((ext_vector_type(8)))  unsigned int v8u;

__device__ __forceinline__ unsigned short f2bf(float f) {
  unsigned int u = __float_as_uint(f);
  u += 0x7FFFu + ((u >> 16) & 1u);          // round-to-nearest-even
  return (unsigned short)(u >> 16);
}
__device__ __forceinline__ float sigm(float x) { return 1.f / (1.f + expf(-x)); }

__device__ __forceinline__ v16bf mk16(uint4 lo, uint4 hi) {
  v8u t;
  t[0]=lo.x; t[1]=lo.y; t[2]=lo.z; t[3]=lo.w;
  t[4]=hi.x; t[5]=hi.y; t[6]=hi.z; t[7]=hi.w;
  return __builtin_bit_cast(v16bf, t);
}

// ---------------- weight convert + transpose:  W[M,K,N] f32 -> Wt[M,N,K] bf16
__global__ void conv_w(const float* __restrict__ W, unsigned short* __restrict__ Wt,
                       int K, int N) {
  size_t idx = (size_t)blockIdx.x * blockDim.x + threadIdx.x;
  size_t tot = (size_t)Mn * K * N;
  if (idx >= tot) return;
  int n = (int)(idx % N);
  size_t r = idx / N;
  int k = (int)(r % K);
  int m = (int)(r / K);
  Wt[((size_t)m * N + n) * K + k] = f2bf(W[idx]);
}

// ---------------- vW[m,r,a] = sum_h v[m,r,h]*Wv[m,h,a]   (step-invariant)
__global__ void vw_k(const float* __restrict__ v, const float* __restrict__ Wv,
                     float* __restrict__ vW) {
  int idx = blockIdx.x * blockDim.x + threadIdx.x;
  if (idx >= Mn * Rn * An) return;
  int a = idx % An; int r = (idx / An) % Rn; int m = idx / (An * Rn);
  float s = 0.f;
  const float* vp = v + ((size_t)m * Rn + r) * Hn;
  const float* wp = Wv + (size_t)m * Hn * An + a;
  for (int h = 0; h < Hn; ++h) s += vp[h] * wp[(size_t)h * An];
  vW[idx] = s;
}

__global__ void vbar_k(const float* __restrict__ v, float* __restrict__ vbar) {
  int idx = blockIdx.x * blockDim.x + threadIdx.x;
  if (idx >= Mn * Hn) return;
  int h = idx % Hn; int m = idx / Hn;
  float s = 0.f;
  for (int r = 0; r < Rn; ++r) s += v[((size_t)m * Rn + r) * Hn + h];
  vbar[idx] = s * (1.f / Rn);
}

// ---------------- fused dual-GEMM via WMMA bf16:
//   out[m, 0..Bact, N] = A1[m]@Wt1^T + (A2?A2[m]@Wt2^T:0) + bias[m]
// A: bf16 [M,16,K] row-major (rows>=Bact zero); Wt: bf16 [M,N,K]
__global__ void gemm_wmma(const unsigned short* __restrict__ A1,
                          const unsigned short* __restrict__ Wt1, int K1,
                          const unsigned short* __restrict__ A2,
                          const unsigned short* __restrict__ Wt2, int K2,
                          const float* __restrict__ bias,
                          float* __restrict__ out, int N, int Bact) {
  int m    = blockIdx.y;
  int wpb  = blockDim.x >> 5;
  int tile = blockIdx.x * wpb + (threadIdx.x >> 5);
  int lane = threadIdx.x & 31;
  int nTiles = (N + 15) >> 4;
  if (tile >= nTiles) return;
  int n0  = tile << 4;
  int col = lane & 15;
  int hi  = lane >> 4;
  int arow  = lane & 15;
  int abase = hi ? 8  : 0;   // A 16-bit layout: lanes 16-31 hold K+8 within 16-group
  int bbase = hi ? 8  : 0;   // B 16-bit layout: lanes 16-31 hold K+8 within 16-group

  v8f c = {};
  float bv = bias ? bias[(size_t)m * N + n0 + col] : 0.f;
#pragma unroll
  for (int i = 0; i < 8; ++i) c[i] = bv;

  {
    const unsigned short* ap = A1 + ((size_t)m * 16 + arow) * K1 + abase;
    const unsigned short* bp = Wt1 + ((size_t)m * N + n0 + col) * K1 + bbase;
    for (int k = 0; k < K1; k += 32) {
      uint4 alo = *(const uint4*)(ap + k);
      uint4 ahi = *(const uint4*)(ap + k + 16);
      uint4 blo = *(const uint4*)(bp + k);
      uint4 bhi = *(const uint4*)(bp + k + 16);
      v16bf av = mk16(alo, ahi);
      v16bf bm = mk16(blo, bhi);
      c = __builtin_amdgcn_wmma_f32_16x16x32_bf16(false, av, false, bm,
                                                  (short)0, c, false, false);
    }
  }
  if (A2) {
    const unsigned short* ap = A2 + ((size_t)m * 16 + arow) * K2 + abase;
    const unsigned short* bp = Wt2 + ((size_t)m * N + n0 + col) * K2 + bbase;
    for (int k = 0; k < K2; k += 32) {
      uint4 alo = *(const uint4*)(ap + k);
      uint4 ahi = *(const uint4*)(ap + k + 16);
      uint4 blo = *(const uint4*)(bp + k);
      uint4 bhi = *(const uint4*)(bp + k + 16);
      v16bf av = mk16(alo, ahi);
      v16bf bm = mk16(blo, bhi);
      c = __builtin_amdgcn_wmma_f32_16x16x32_bf16(false, av, false, bm,
                                                  (short)0, c, false, false);
    }
  }
  float* o = out + (size_t)m * Bn * N;
#pragma unroll
  for (int i = 0; i < 8; ++i) {
    int r = i + (hi ? 8 : 0);                 // C/D layout: VGPR i -> M=i (+8 for hi lanes)
    if (r < Bact) o[(size_t)r * N + n0 + col] = c[i];
  }
}

// ---------------- LSTM pointwise: gates g[M,Bn,4H], cprev -> h,c
__global__ void lstm_k(const float* __restrict__ g, const float* __restrict__ cprev,
                       float* __restrict__ hout, float* __restrict__ cout, int Bact) {
  int idx = blockIdx.x * blockDim.x + threadIdx.x;
  if (idx >= Mn * Bact * Hn) return;
  int h = idx % Hn; int b = (idx / Hn) % Bact; int m = idx / (Hn * Bact);
  const float* gr = g + ((size_t)m * Bn + b) * 4 * Hn;
  float ii = sigm(gr[h]);
  float ff = sigm(gr[Hn + h]);
  float gg = tanhf(gr[2 * Hn + h]);
  float oo = sigm(gr[3 * Hn + h]);
  size_t s = ((size_t)m * Bn + b) * Hn + h;
  float cn = ff * cprev[s] + ii * gg;
  cout[s] = cn;
  hout[s] = oo * tanhf(cn);
}

// ---------------- attention: att=tanh(vW+haw)@wa, softmax_R, vhat=alpha@v
__global__ void attn_k(const float* __restrict__ vW, const float* __restrict__ haw,
                       const float* __restrict__ wa, const float* __restrict__ v,
                       float* __restrict__ vhat) {
  int b = blockIdx.x, m = blockIdx.y, tid = threadIdx.x;   // 256 threads
  __shared__ float att[Rn];
  __shared__ float red[256];
  const float* vWm  = vW + (size_t)m * Rn * An;
  const float* hawb = haw + ((size_t)m * Bn + b) * An;
  const float* wam  = wa + (size_t)m * An;
  for (int r = 0; r < Rn; ++r) {
    float s = 0.f;
    for (int a = tid; a < An; a += 256)
      s += tanhf(vWm[r * An + a] + hawb[a]) * wam[a];
    red[tid] = s; __syncthreads();
    for (int off = 128; off; off >>= 1) { if (tid < off) red[tid] += red[tid + off]; __syncthreads(); }
    if (tid == 0) att[r] = red[0];
    __syncthreads();
  }
  if (tid == 0) {
    float mx = att[0];
    for (int r = 1; r < Rn; ++r) mx = fmaxf(mx, att[r]);
    float se = 0.f;
    for (int r = 0; r < Rn; ++r) { att[r] = expf(att[r] - mx); se += att[r]; }
    float inv = 1.f / se;
    for (int r = 0; r < Rn; ++r) att[r] *= inv;
  }
  __syncthreads();
  const float* vm = v + (size_t)m * Rn * Hn;
  float* o = vhat + ((size_t)m * Bn + b) * Hn;
  for (int h = tid; h < Hn; h += 256) {
    float s = 0.f;
    for (int r = 0; r < Rn; ++r) s += att[r] * vm[r * Hn + h];
    o[h] = s;
  }
}

// ---------------- activation packing to bf16 [M,16,K]
__global__ void pack_x1(const float* __restrict__ h2, const float* __restrict__ vbar,
                        const float* __restrict__ E, const int* __restrict__ toks,
                        unsigned short* __restrict__ out, int Bact) {
  const int K = 2 * Hn + En;
  int idx = blockIdx.x * blockDim.x + threadIdx.x;
  if (idx >= Mn * 16 * K) return;
  int c = idx % K; int row = (idx / K) % 16; int m = idx / (K * 16);
  float val = 0.f;
  if (row < Bact) {
    if (c < Hn)            val = h2[((size_t)m * Bn + row) * Hn + c];
    else if (c < 2 * Hn)   val = vbar[(size_t)m * Hn + (c - Hn)];
    else                   val = E[((size_t)m * Vn + toks[row]) * (size_t)En + (c - 2 * Hn)];
  }
  out[idx] = f2bf(val);
}
__global__ void pack_x2(const float* __restrict__ vhat, const float* __restrict__ h1,
                        unsigned short* __restrict__ out, int Bact) {
  const int K = 2 * Hn;
  int idx = blockIdx.x * blockDim.x + threadIdx.x;
  if (idx >= Mn * 16 * K) return;
  int c = idx % K; int row = (idx / K) % 16; int m = idx / (K * 16);
  float val = 0.f;
  if (row < Bact) {
    if (c < Hn) val = vhat[((size_t)m * Bn + row) * Hn + c];
    else        val = h1[((size_t)m * Bn + row) * Hn + (c - Hn)];
  }
  out[idx] = f2bf(val);
}
__global__ void pack_h(const float* __restrict__ s, unsigned short* __restrict__ out, int Bact) {
  int idx = blockIdx.x * blockDim.x + threadIdx.x;
  if (idx >= Mn * 16 * Hn) return;
  int c = idx % Hn; int row = (idx / Hn) % 16; int m = idx / (Hn * 16);
  float val = (row < Bact) ? s[((size_t)m * Bn + row) * Hn + c] : 0.f;
  out[idx] = f2bf(val);
}

// ---------------- per-(m,b) log-sum-exp stats
__global__ void lse_k(const float* __restrict__ logits, float* __restrict__ mx,
                      float* __restrict__ lse) {
  int b = blockIdx.x, m = blockIdx.y, tid = threadIdx.x;   // 256 threads
  const float* l = logits + ((size_t)m * Bn + b) * Vn;
  __shared__ float red[256];
  float mv = -3.4e38f;
  for (int i = tid; i < Vn; i += 256) mv = fmaxf(mv, l[i]);
  red[tid] = mv; __syncthreads();
  for (int o = 128; o; o >>= 1) { if (tid < o) red[tid] = fmaxf(red[tid], red[tid + o]); __syncthreads(); }
  float m0 = red[0]; __syncthreads();
  float s = 0.f;
  for (int i = tid; i < Vn; i += 256) s += expf(l[i] - m0);
  red[tid] = s; __syncthreads();
  for (int o = 128; o; o >>= 1) { if (tid < o) red[tid] += red[tid + o]; __syncthreads(); }
  if (tid == 0) { mx[m * Bn + b] = m0; lse[m * Bn + b] = logf(red[0]); }
}

// ---------------- step-0 total over V (single <start> beam)
__global__ void total0_k(const float* __restrict__ logits, const float* __restrict__ mx,
                         const float* __restrict__ lse, float* __restrict__ total) {
  int v = blockIdx.x * blockDim.x + threadIdx.x;
  if (v >= Vn) return;
  float s = 0.f;
  for (int m = 0; m < Mn; ++m)
    s += logits[((size_t)m * Bn) * Vn + v] - mx[m * Bn] - lse[m * Bn];
  float lp = s * (1.f / Mn);
  if (v == PAD_T || v == START_T) lp += NEGV;
  total[v] = lp;
}

// ---------------- step-t total over B*V
__global__ void total_k(const float* __restrict__ logits, const float* __restrict__ mx,
                        const float* __restrict__ lse, const float* __restrict__ mask,
                        const float* __restrict__ scores, const int* __restrict__ toks,
                        const int* __restrict__ fin, float* __restrict__ total) {
  int idx = blockIdx.x * blockDim.x + threadIdx.x;
  if (idx >= Bn * Vn) return;
  int v = idx % Vn; int b = idx / Vn;
  float lp;
  if (fin[b]) {
    lp = (v == PAD_T) ? 0.f : NEGV;
  } else {
    float s = 0.f;
    for (int m = 0; m < Mn; ++m)
      s += logits[((size_t)m * Bn + b) * Vn + v] - mx[m * Bn + b] - lse[m * Bn + b];
    lp = s * (1.f / Mn) + mask[(size_t)b * Vn + v];
    int t = toks[b];
    if (v == END_T && t >= BADLO && t < BADHI) lp += NEGV;
  }
  total[idx] = scores[b] + lp;
}

// ---------------- top-5 over n elements (single block, 256 threads)
__global__ void topk_k(const float* __restrict__ x, int n,
                       float* __restrict__ outv, int* __restrict__ outi) {
  __shared__ float sv[256 * 5];
  __shared__ int   si[256 * 5];
  int tid = threadIdx.x;
  float bv[5]; int bi[5];
  for (int j = 0; j < 5; ++j) { bv[j] = -3.4e38f; bi[j] = 0; }
  for (int i = tid; i < n; i += 256) {
    float val = x[i];
    if (val > bv[4]) {
      int j = 4;
      while (j > 0 && bv[j - 1] < val) { bv[j] = bv[j - 1]; bi[j] = bi[j - 1]; --j; }
      bv[j] = val; bi[j] = i;
    }
  }
  for (int j = 0; j < 5; ++j) { sv[tid * 5 + j] = bv[j]; si[tid * 5 + j] = bi[j]; }
  __syncthreads();
  if (tid == 0) {
    float fv[5]; int fi[5];
    for (int j = 0; j < 5; ++j) { fv[j] = -3.4e38f; fi[j] = 0; }
    for (int i = 0; i < 256 * 5; ++i) {
      float val = sv[i];
      if (val > fv[4]) {
        int id = si[i], j = 4;
        while (j > 0 && fv[j - 1] < val) { fv[j] = fv[j - 1]; fi[j] = fi[j - 1]; --j; }
        fv[j] = val; fi[j] = id;
      }
    }
    for (int j = 0; j < 5; ++j) { outv[j] = fv[j]; outi[j] = fi[j]; }
  }
}

// ---------------- step-0 state init (beam 0 only)
__global__ void init0_k(const float* __restrict__ h0, const float* __restrict__ c0,
                        float* h1c, float* c1c, float* h2c, float* c2c, int* toks) {
  int idx = blockIdx.x * blockDim.x + threadIdx.x;
  if (idx >= Mn * Hn) return;
  int h = idx % Hn; int m = idx / Hn;
  size_t s = (size_t)m * Bn * Hn + h;      // beam 0 row
  h1c[s] = 0.f; c1c[s] = 0.f;
  h2c[s] = h0[idx]; c2c[s] = c0[idx];
  if (idx == 0) toks[0] = START_T;
}

// ---------------- step-0 beam creation (tokens/scores/finished/seqs/mask)
__global__ void beams0_k(const float* __restrict__ topv, const int* __restrict__ topi,
                         int* toks, float* scores, int* fin, int* seq, float* mask) {
  int idx = blockIdx.x * blockDim.x + threadIdx.x;
  if (idx >= Bn * Vn) return;
  int v = idx % Vn; int b = idx / Vn;
  int tok = topi[b];
  float mv = (v == PAD_T || v == START_T) ? NEGV : 0.f;
  int stop = (tok >= STOPLO && tok < STOPHI);
  if (!stop && v == tok) mv += NEGV;
  mask[idx] = mv;
  if (v == 0) { toks[b] = tok; scores[b] = topv[b]; fin[b] = (tok == END_T); }
  if (v < Ln) seq[b * Ln + v] = (v == 0) ? tok : 0;
}

__global__ void bcast_k(const float* __restrict__ src, float* __restrict__ dst) {
  int idx = blockIdx.x * blockDim.x + threadIdx.x;
  if (idx >= Mn * Bn * Hn) return;
  int h = idx % Hn; int m = idx / (Bn * Hn);
  dst[idx] = src[(size_t)m * Bn * Hn + h];   // beam 0 -> all beams
}

// ---------------- beam bookkeeping after top-k (step >= 1)
__global__ void beam_update_k(const float* __restrict__ topv, const int* __restrict__ topi,
                              const int* __restrict__ fin_old, int* fin_new, int* toks,
                              float* scores, int* parents, int* tokw, int* ns) {
  if (threadIdx.x == 0 && blockIdx.x == 0) {
    int po[Bn], nt[Bn], pf[Bn];
    for (int b = 0; b < Bn; ++b) { int f = topi[b]; po[b] = f / Vn; nt[b] = f % Vn; }
    for (int b = 0; b < Bn; ++b) pf[b] = fin_old[po[b]];
    for (int b = 0; b < Bn; ++b) {
      parents[b] = po[b];
      toks[b]    = nt[b];
      scores[b]  = topv[b];
      tokw[b]    = pf[b] ? PAD_T : nt[b];
      fin_new[b] = pf[b] || (nt[b] == END_T);
      int stop   = (nt[b] >= STOPLO && nt[b] < STOPHI);
      ns[b]      = (!stop) && (!pf[b]);
    }
  }
}

__global__ void seq_gather_k(const int* __restrict__ seq_old, int* __restrict__ seq_new,
                             const int* __restrict__ parents, const int* __restrict__ tokw, int t) {
  int idx = blockIdx.x * blockDim.x + threadIdx.x;
  if (idx >= Bn * Ln) return;
  int j = idx % Ln; int b = idx / Ln;
  seq_new[idx] = (j == t) ? tokw[b] : seq_old[parents[b] * Ln + j];
}

__global__ void mask_gather_k(const float* __restrict__ mold, float* __restrict__ mnew,
                              const int* __restrict__ parents, const int* __restrict__ toks,
                              const int* __restrict__ ns) {
  int idx = blockIdx.x * blockDim.x + threadIdx.x;
  if (idx >= Bn * Vn) return;
  int v = idx % Vn; int b = idx / Vn;
  float mv = mold[(size_t)parents[b] * Vn + v];
  if (ns[b] && v == toks[b]) mv += NEGV;
  mnew[idx] = mv;
}

__global__ void state_gather_k(const float* __restrict__ src, float* __restrict__ dst,
                               const int* __restrict__ parents) {
  int idx = blockIdx.x * blockDim.x + threadIdx.x;
  if (idx >= Mn * Bn * Hn) return;
  int h = idx % Hn; int b = (idx / Hn) % Bn; int m = idx / (Hn * Bn);
  dst[idx] = src[((size_t)m * Bn + parents[b]) * Hn + h];
}

__global__ void final_k(const float* __restrict__ scores, const int* __restrict__ seq,
                        float* __restrict__ out) {
  int idx = blockIdx.x * blockDim.x + threadIdx.x;
  if (idx >= Bn + Bn * Ln) return;
  if (idx < Bn) out[idx] = scores[idx];
  else ((int*)out)[idx] = seq[idx - Bn];
}

// =====================================================================
extern "C" void kernel_launch(void* const* d_in, const int* in_sizes, int n_in,
                              void* d_out, int out_size, void* d_ws, size_t ws_size,
                              hipStream_t stream) {
  const float* E   = (const float*)d_in[0];
  const float* Wx1 = (const float*)d_in[1];
  const float* Wh1 = (const float*)d_in[2];
  const float* b1  = (const float*)d_in[3];
  const float* Wv  = (const float*)d_in[4];
  const float* Wha = (const float*)d_in[5];
  const float* wa  = (const float*)d_in[6];
  const float* Wx2 = (const float*)d_in[7];
  const float* Wh2 = (const float*)d_in[8];
  const float* b2  = (const float*)d_in[9];
  const float* Wo  = (const float*)d_in[10];
  const float* bo  = (const float*)d_in[11];
  const float* v   = (const float*)d_in[12];
  const float* h0  = (const float*)d_in[13];
  const float* c0  = (const float*)d_in[14];
  (void)in_sizes; (void)n_in; (void)out_size; (void)ws_size;

  char* wp = (char*)d_ws;
  auto carve = [&](size_t bytes) -> char* {
    char* r = wp; wp += (bytes + 255) & ~(size_t)255; return r;
  };
  const int K1 = 2 * Hn + En;   // 1536
  const int K2 = 2 * Hn;        // 1024
  const int G  = 4 * Hn;        // 2048 gates

  unsigned short* WtX1 = (unsigned short*)carve((size_t)Mn * G * K1 * 2);
  unsigned short* WtH1 = (unsigned short*)carve((size_t)Mn * G * Hn * 2);
  unsigned short* WtHA = (unsigned short*)carve((size_t)Mn * An * Hn * 2);
  unsigned short* WtX2 = (unsigned short*)carve((size_t)Mn * G * K2 * 2);
  unsigned short* WtH2 = (unsigned short*)carve((size_t)Mn * G * Hn * 2);
  unsigned short* WtWo = (unsigned short*)carve((size_t)Mn * Vn * Hn * 2);
  float* vW   = (float*)carve((size_t)Mn * Rn * An * 4);
  float* vbar = (float*)carve((size_t)Mn * Hn * 4);
  float* h1c = (float*)carve((size_t)Mn * Bn * Hn * 4);
  float* c1c = (float*)carve((size_t)Mn * Bn * Hn * 4);
  float* h2c = (float*)carve((size_t)Mn * Bn * Hn * 4);
  float* c2c = (float*)carve((size_t)Mn * Bn * Hn * 4);
  float* h1n = (float*)carve((size_t)Mn * Bn * Hn * 4);
  float* c1n = (float*)carve((size_t)Mn * Bn * Hn * 4);
  float* h2n = (float*)carve((size_t)Mn * Bn * Hn * 4);
  float* c2n = (float*)carve((size_t)Mn * Bn * Hn * 4);
  unsigned short* Ax1  = (unsigned short*)carve((size_t)Mn * 16 * K1 * 2);
  unsigned short* Ah1p = (unsigned short*)carve((size_t)Mn * 16 * Hn * 2);
  unsigned short* Ah1n = (unsigned short*)carve((size_t)Mn * 16 * Hn * 2);
  unsigned short* Ax2  = (unsigned short*)carve((size_t)Mn * 16 * K2 * 2);
  unsigned short* Ah2p = (unsigned short*)carve((size_t)Mn * 16 * Hn * 2);
  unsigned short* Ah2o = (unsigned short*)carve((size_t)Mn * 16 * Hn * 2);
  float* g1     = (float*)carve((size_t)Mn * Bn * G * 4);
  float* g2     = (float*)carve((size_t)Mn * Bn * G * 4);
  float* haw    = (float*)carve((size_t)Mn * Bn * An * 4);
  float* vhat   = (float*)carve((size_t)Mn * Bn * Hn * 4);
  float* logits = (float*)carve((size_t)Mn * Bn * Vn * 4);
  float* mxb    = (float*)carve((size_t)Mn * Bn * 4);
  float* lseb   = (float*)carve((size_t)Mn * Bn * 4);
  float* total  = (float*)carve((size_t)Bn * Vn * 4);
  float* maskA  = (float*)carve((size_t)Bn * Vn * 4);
  float* maskB  = (float*)carve((size_t)Bn * Vn * 4);
  float* scores = (float*)carve(Bn * 4);
  int* toks     = (int*)carve(Bn * 4);
  int* finA     = (int*)carve(Bn * 4);
  int* finB     = (int*)carve(Bn * 4);
  int* parents  = (int*)carve(Bn * 4);
  int* tokw     = (int*)carve(Bn * 4);
  int* nsb      = (int*)carve(Bn * 4);
  int* seqA     = (int*)carve(Bn * Ln * 4);
  int* seqB     = (int*)carve(Bn * Ln * 4);
  float* topv   = (float*)carve(Bn * 4);
  int* topi     = (int*)carve(Bn * 4);

  auto blks = [](size_t tot, int t) { return dim3((unsigned)((tot + t - 1) / t)); };

  // ---- one-time (per call) precompute: bf16 weight transposes, vW, vbar
  conv_w<<<blks((size_t)Mn * K1 * G, 256), 256, 0, stream>>>(Wx1, WtX1, K1, G);
  conv_w<<<blks((size_t)Mn * Hn * G, 256), 256, 0, stream>>>(Wh1, WtH1, Hn, G);
  conv_w<<<blks((size_t)Mn * Hn * An, 256), 256, 0, stream>>>(Wha, WtHA, Hn, An);
  conv_w<<<blks((size_t)Mn * K2 * G, 256), 256, 0, stream>>>(Wx2, WtX2, K2, G);
  conv_w<<<blks((size_t)Mn * Hn * G, 256), 256, 0, stream>>>(Wh2, WtH2, Hn, G);
  conv_w<<<blks((size_t)Mn * Hn * Vn, 256), 256, 0, stream>>>(Wo, WtWo, Hn, Vn);
  vw_k  <<<blks((size_t)Mn * Rn * An, 256), 256, 0, stream>>>(v, Wv, vW);
  vbar_k<<<blks((size_t)Mn * Hn, 256), 256, 0, stream>>>(v, vbar);

  auto gemm = [&](const unsigned short* A1p, const unsigned short* W1p, int k1,
                  const unsigned short* A2p, const unsigned short* W2p, int k2,
                  const float* bias, float* out, int N, int Bact) {
    int tiles = (N + 15) / 16;
    dim3 grd((tiles + 3) / 4, Mn), blk(128);   // 4 waves/block, one 16-col tile/wave
    gemm_wmma<<<grd, blk, 0, stream>>>(A1p, W1p, k1, A2p, W2p, k2, bias, out, N, Bact);
  };

  auto run_step = [&](int Bact) {
    pack_x1<<<blks((size_t)Mn * 16 * K1, 256), 256, 0, stream>>>(h2c, vbar, E, toks, Ax1, Bact);
    pack_h <<<blks((size_t)Mn * 16 * Hn, 256), 256, 0, stream>>>(h1c, Ah1p, Bact);
    gemm(Ax1, WtX1, K1, Ah1p, WtH1, Hn, b1, g1, G, Bact);
    lstm_k<<<blks((size_t)Mn * Bact * Hn, 256), 256, 0, stream>>>(g1, c1c, h1n, c1n, Bact);
    pack_h<<<blks((size_t)Mn * 16 * Hn, 256), 256, 0, stream>>>(h1n, Ah1n, Bact);
    gemm(Ah1n, WtHA, Hn, nullptr, nullptr, 0, nullptr, haw, An, Bact);
    attn_k<<<dim3(Bact, Mn), 256, 0, stream>>>(vW, haw, wa, v, vhat);
    pack_x2<<<blks((size_t)Mn * 16 * K2, 256), 256, 0, stream>>>(vhat, h1n, Ax2, Bact);
    pack_h <<<blks((size_t)Mn * 16 * Hn, 256), 256, 0, stream>>>(h2c, Ah2p, Bact);
    gemm(Ax2, WtX2, K2, Ah2p, WtH2, Hn, b2, g2, G, Bact);
    lstm_k<<<blks((size_t)Mn * Bact * Hn, 256), 256, 0, stream>>>(g2, c2c, h2n, c2n, Bact);
    pack_h<<<blks((size_t)Mn * 16 * Hn, 256), 256, 0, stream>>>(h2n, Ah2o, Bact);
    gemm(Ah2o, WtWo, Hn, nullptr, nullptr, 0, bo, logits, Vn, Bact);
    lse_k<<<dim3(Bact, Mn), 256, 0, stream>>>(logits, mxb, lseb);
  };

  // -------- step 0: single <start> beam --------
  init0_k<<<blks((size_t)Mn * Hn, 256), 256, 0, stream>>>(h0, c0, h1c, c1c, h2c, c2c, toks);
  run_step(1);
  total0_k<<<blks(Vn, 256), 256, 0, stream>>>(logits, mxb, lseb, total);
  topk_k<<<1, 256, 0, stream>>>(total, Vn, topv, topi);
  beams0_k<<<blks((size_t)Bn * Vn, 256), 256, 0, stream>>>(topv, topi, toks, scores, finA, seqA, maskA);
  bcast_k<<<blks((size_t)Mn * Bn * Hn, 256), 256, 0, stream>>>(h1n, h1c);
  bcast_k<<<blks((size_t)Mn * Bn * Hn, 256), 256, 0, stream>>>(c1n, c1c);
  bcast_k<<<blks((size_t)Mn * Bn * Hn, 256), 256, 0, stream>>>(h2n, h2c);
  bcast_k<<<blks((size_t)Mn * Bn * Hn, 256), 256, 0, stream>>>(c2n, c2c);

  float* maskCur = maskA; float* maskNxt = maskB;
  int* seqCur = seqA;     int* seqNxt = seqB;
  int* finCur = finA;     int* finNxt = finB;

  // -------- steps 1..29 --------
  for (int t = 1; t < Ln; ++t) {
    run_step(Bn);
    total_k<<<blks((size_t)Bn * Vn, 256), 256, 0, stream>>>(logits, mxb, lseb, maskCur,
                                                            scores, toks, finCur, total);
    topk_k<<<1, 256, 0, stream>>>(total, Bn * Vn, topv, topi);
    beam_update_k<<<1, 32, 0, stream>>>(topv, topi, finCur, finNxt, toks, scores,
                                        parents, tokw, nsb);
    seq_gather_k<<<blks((size_t)Bn * Ln, 256), 256, 0, stream>>>(seqCur, seqNxt, parents, tokw, t);
    mask_gather_k<<<blks((size_t)Bn * Vn, 256), 256, 0, stream>>>(maskCur, maskNxt, parents, toks, nsb);
    state_gather_k<<<blks((size_t)Mn * Bn * Hn, 256), 256, 0, stream>>>(h1n, h1c, parents);
    state_gather_k<<<blks((size_t)Mn * Bn * Hn, 256), 256, 0, stream>>>(c1n, c1c, parents);
    state_gather_k<<<blks((size_t)Mn * Bn * Hn, 256), 256, 0, stream>>>(h2n, h2c, parents);
    state_gather_k<<<blks((size_t)Mn * Bn * Hn, 256), 256, 0, stream>>>(c2n, c2c, parents);
    { float* tm = maskCur; maskCur = maskNxt; maskNxt = tm; }
    { int* ts = seqCur; seqCur = seqNxt; seqNxt = ts; }
    { int* tf = finCur; finCur = finNxt; finNxt = tf; }
  }

  final_k<<<1, 256, 0, stream>>>(scores, seqCur, (float*)d_out);
}
